// Indexer_29085518528942
// MI455X (gfx1250) — compile-verified, matrix operand order checked by reading, not run on hardware
//
#include <hip/hip_runtime.h>
#include <hip/hip_bf16.h>

#define SQ   4096
#define NH   32
#define DH   128
#define RR   64
#define QLR  1536
#define HIDN 2048
#define HD   4096   // NH*DH
#define TOPK 2048
#define NEGV (-1e30f)

typedef __bf16 v16bf __attribute__((ext_vector_type(16)));
typedef float  v8f   __attribute__((ext_vector_type(8)));

union FragU { v16bf v; uint4 u[2]; };

__device__ __forceinline__ unsigned short f2bf(float f) {
    unsigned int u = __float_as_uint(f);
    u = (u + 0x7FFFu + ((u >> 16) & 1u)) >> 16;
    return (unsigned short)u;
}

__device__ __forceinline__ v8f wmma_bf16(v16bf a, v16bf b, v8f c) {
    return __builtin_amdgcn_wmma_f32_16x16x32_bf16(false, a, false, b, (short)0, c, false, false);
}

// ---- CDNA5 async global->LDS copy (ASYNCcnt-tracked, no VGPR round trip) ----
// Per-lane: 16 bytes from gSrc[lane] into LDS at ldsDst[lane].
__device__ __forceinline__ void async_copy_b128(void* ldsDst, const void* gSrc) {
    unsigned ldsoff = (unsigned)(unsigned long long)ldsDst;  // low 32 bits = LDS offset
    asm volatile("global_load_async_to_lds_b128 %0, %1, off"
                 :: "v"(ldsoff), "v"(gSrc) : "memory");
}
__device__ __forceinline__ void wait_async() {
    asm volatile("s_wait_asynccnt 0x0" ::: "memory");
}

// Load one 16x32 bf16 fragment (A or Bt layout, K contiguous per row) from LDS.
__device__ __forceinline__ v16bf load_frag(const unsigned short* lds, int rowBase,
                                           int strideHalves, int kOff) {
    int lane = threadIdx.x & 31;
    int r  = rowBase + (lane & 15);
    int kb = kOff + ((lane >> 4) << 3);
    FragU f;
    f.u[0] = *(const uint4*)(lds + r * strideHalves + kb);
    f.u[1] = *(const uint4*)(lds + r * strideHalves + kb + 16);
    return f.v;
}

// ---- RoPE (interleaved, first 64 dims) on 4-consecutive-per-lane layout ----
__device__ __forceinline__ void rope4(float x[4], int pos) {
    int lane = threadIdx.x & 31;
    if (lane < 16) {
        #pragma unroll
        for (int p = 0; p < 2; ++p) {
            int pairIdx = lane * 2 + p;               // 0..31
            float freq = __expf(-(float)pairIdx * (9.210340371976184f / 32.0f));
            float ang = (float)pos * freq;
            float s, c;
            __sincosf(ang, &s, &c);
            float xe = x[2 * p], xo = x[2 * p + 1];
            x[2 * p]     = xe * c - xo * s;
            x[2 * p + 1] = xe * s + xo * c;
        }
    }
}

// ---- FWHT over 128 elems: lane t holds elems [4t..4t+3]; scaled by 128^-0.5 ----
__device__ __forceinline__ void fwht128(float x[4]) {
    float a, b;
    a = x[0]; b = x[1]; x[0] = a + b; x[1] = a - b;   // h=1
    a = x[2]; b = x[3]; x[2] = a + b; x[3] = a - b;
    a = x[0]; b = x[2]; x[0] = a + b; x[2] = a - b;   // h=2
    a = x[1]; b = x[3]; x[1] = a + b; x[3] = a - b;
    int lane = threadIdx.x & 31;
    #pragma unroll
    for (int m = 1; m <= 16; m <<= 1) {               // h=4..64 cross-lane
        bool upper = (lane & m) != 0;
        #pragma unroll
        for (int j = 0; j < 4; ++j) {
            float v = __shfl_xor(x[j], m, 32);
            x[j] = upper ? (v - x[j]) : (x[j] + v);
        }
    }
    const float s = 0.08838834764831845f;             // 128^-0.5
    #pragma unroll
    for (int j = 0; j < 4; ++j) x[j] *= s;
}

// ======================= conversion kernels =======================
__global__ void cast_f32_bf16(const float* __restrict__ src,
                              unsigned short* __restrict__ dst, long n) {
    long i = (long)blockIdx.x * blockDim.x + threadIdx.x;
    if (i < n) dst[i] = f2bf(src[i]);
}

// src rows x cols (row-major) -> dst cols x rows (row-major), bf16
__global__ void transpose_cast(const float* __restrict__ src,
                               unsigned short* __restrict__ dst,
                               int rows, int cols) {
    long i = (long)blockIdx.x * blockDim.x + threadIdx.x;
    long n = (long)rows * cols;
    if (i < n) {
        int r = (int)(i / cols), c = (int)(i % cols);
        dst[(long)c * rows + r] = f2bf(src[i]);
    }
}

// ============ generic bf16 GEMM: C(MxN) = A(MxK) * Bt(NxK)^T ============
// BM=64 BN=64 BK=32, 256 threads = 8 waves (2x4), wave tile 32x16.
__global__ void gemm_bf16_nt(const unsigned short* __restrict__ A,
                             const unsigned short* __restrict__ Bt,
                             float* __restrict__ C, int M, int N, int K) {
    __shared__ unsigned short As[64 * 40];
    __shared__ unsigned short Bs[64 * 40];
    int m0 = blockIdx.y * 64, n0 = blockIdx.x * 64;
    int tid = threadIdx.x, wid = tid >> 5, lane = tid & 31;
    int wm = wid >> 2, wn = wid & 3;
    v8f acc[2] = {};
    int r = tid >> 2, kc = (tid & 3) * 8;
    for (int k0 = 0; k0 < K; k0 += 32) {
        async_copy_b128(As + r * 40 + kc, A + (long)(m0 + r) * K + k0 + kc);
        uint4 bv = {0, 0, 0, 0};
        if (n0 + r < N) bv = *(const uint4*)(Bt + (long)(n0 + r) * K + k0 + kc);
        *(uint4*)(Bs + r * 40 + kc) = bv;
        if (k0 + 32 < K)
            __builtin_prefetch(A + (long)(m0 + r) * K + k0 + 32 + kc, 0, 0);
        wait_async();
        __syncthreads();
        v16bf a0 = load_frag(As, wm * 32, 40, 0);
        v16bf a1 = load_frag(As, wm * 32 + 16, 40, 0);
        v16bf b0 = load_frag(Bs, wn * 16, 40, 0);
        acc[0] = wmma_bf16(a0, b0, acc[0]);
        acc[1] = wmma_bf16(a1, b0, acc[1]);
        __syncthreads();
    }
    #pragma unroll
    for (int mt = 0; mt < 2; ++mt)
        #pragma unroll
        for (int g = 0; g < 8; ++g) {
            int row = m0 + wm * 32 + mt * 16 + g + ((lane >> 4) << 3);
            int col = n0 + wn * 16 + (lane & 15);
            if (col < N) C[(long)row * N + col] = acc[mt][g];
        }
}

// ====== post: kw (Sx160 f32) -> k bf16 (LN+RoPE+FWHT), w f32 (scaled) ======
__global__ void kw_post(const float* __restrict__ KW,
                        const float* __restrict__ gamma,
                        const float* __restrict__ beta,
                        const int* __restrict__ positions,
                        unsigned short* __restrict__ Kb,
                        float* __restrict__ Wf) {
    int wid = threadIdx.x >> 5, lane = threadIdx.x & 31;
    int row = blockIdx.x * 8 + wid;
    const float* src = KW + (long)row * 160;
    float x[4];
    *(float4*)x = *(const float4*)(src + lane * 4);
    float wv = src[128 + lane];
    // LayerNorm over 128
    float s = x[0] + x[1] + x[2] + x[3];
    #pragma unroll
    for (int m = 1; m <= 16; m <<= 1) s += __shfl_xor(s, m, 32);
    float mu = s * (1.0f / 128.0f);
    float v = 0.f;
    #pragma unroll
    for (int j = 0; j < 4; ++j) { float d = x[j] - mu; v += d * d; }
    #pragma unroll
    for (int m = 1; m <= 16; m <<= 1) v += __shfl_xor(v, m, 32);
    float inv = rsqrtf(v * (1.0f / 128.0f) + 1e-6f);
    #pragma unroll
    for (int j = 0; j < 4; ++j)
        x[j] = (x[j] - mu) * inv * gamma[lane * 4 + j] + beta[lane * 4 + j];
    rope4(x, positions[row]);
    fwht128(x);
    ushort4 o;
    o.x = f2bf(x[0]); o.y = f2bf(x[1]); o.z = f2bf(x[2]); o.w = f2bf(x[3]);
    *(ushort4*)(Kb + (long)row * 128 + lane * 4) = o;
    Wf[(long)row * 32 + lane] = wv * 0.015625f;  // (H*D)^-0.5 = 1/64
}

// ====== q GEMM (Sx1536 @ 1536x4096) fused with RoPE+FWHT -> bf16 q ======
// BM=64 BN=128(= one head) BK=32, 8 waves (2x4), wave tile 32x32.
__global__ void qgemm_rope_fwht(const unsigned short* __restrict__ A,
                                const unsigned short* __restrict__ Bt,
                                const int* __restrict__ positions,
                                unsigned short* __restrict__ Q) {
    __shared__ __align__(16) char smem[64 * 128 * 4];   // 32 KB, aliased
    unsigned short* As = (unsigned short*)smem;          // 64 x 40 halves
    unsigned short* Bs = (unsigned short*)(smem + 64 * 40 * 2); // 128 x 40
    int head = blockIdx.x, n0 = head * 128, m0 = blockIdx.y * 64;
    int tid = threadIdx.x, wid = tid >> 5, lane = tid & 31;
    int wm = wid >> 2, wn = wid & 3;
    v8f acc[2][2] = {};
    int r = tid >> 2, kc = (tid & 3) * 8;
    for (int k0 = 0; k0 < QLR; k0 += 32) {
        async_copy_b128(As + r * 40 + kc, A + (long)(m0 + r) * QLR + k0 + kc);
        #pragma unroll
        for (int it = 0; it < 2; ++it) {
            int rb = it * 64 + r;
            async_copy_b128(Bs + rb * 40 + kc, Bt + (long)(n0 + rb) * QLR + k0 + kc);
        }
        if (k0 + 32 < QLR)
            __builtin_prefetch(Bt + (long)(n0 + r) * QLR + k0 + 32 + kc, 0, 0);
        wait_async();
        __syncthreads();
        v16bf a0 = load_frag(As, wm * 32, 40, 0);
        v16bf a1 = load_frag(As, wm * 32 + 16, 40, 0);
        v16bf b0 = load_frag(Bs, wn * 32, 40, 0);
        v16bf b1 = load_frag(Bs, wn * 32 + 16, 40, 0);
        acc[0][0] = wmma_bf16(a0, b0, acc[0][0]);
        acc[0][1] = wmma_bf16(a0, b1, acc[0][1]);
        acc[1][0] = wmma_bf16(a1, b0, acc[1][0]);
        acc[1][1] = wmma_bf16(a1, b1, acc[1][1]);
        __syncthreads();
    }
    // dump accumulators to LDS as 64x128 f32 tile
    float* Cs = (float*)smem;
    #pragma unroll
    for (int mt = 0; mt < 2; ++mt)
        #pragma unroll
        for (int nt = 0; nt < 2; ++nt)
            #pragma unroll
            for (int g = 0; g < 8; ++g) {
                int row = wm * 32 + mt * 16 + g + ((lane >> 4) << 3);
                int col = wn * 32 + nt * 16 + (lane & 15);
                Cs[row * 128 + col] = acc[mt][nt][g];
            }
    __syncthreads();
    // epilogue: wave wid handles rows wid*8 .. wid*8+7 ; RoPE + FWHT -> bf16
    for (int rr = 0; rr < 8; ++rr) {
        int row = wid * 8 + rr;
        int s = m0 + row;
        float x[4];
        *(float4*)x = *(const float4*)(Cs + row * 128 + lane * 4);
        rope4(x, positions[s]);
        fwht128(x);
        ushort4 o;
        o.x = f2bf(x[0]); o.y = f2bf(x[1]); o.z = f2bf(x[2]); o.w = f2bf(x[3]);
        *(ushort4*)(Q + (long)s * HD + head * 128 + lane * 4) = o;
    }
}

// ====== logits: sum_h w[i,h]*relu(q_h @ k^T), causal mask, f32 out ======
// Block tile: 64 rows(i) x 128 cols(j); k-tile resident in LDS across 32 heads.
__global__ void logits_kernel(const unsigned short* __restrict__ Q,
                              const unsigned short* __restrict__ Kb,
                              const float* __restrict__ W,
                              float* __restrict__ out) {
    int i0 = blockIdx.y * 64, j0 = blockIdx.x * 128;
    int tid = threadIdx.x, wid = tid >> 5, lane = tid & 31;
    if (j0 > i0 + 63) {                       // fully above the diagonal
        #pragma unroll 4
        for (int it = 0; it < 32; ++it) {
            int idx = it * 256 + tid;
            int r = idx >> 7, c = idx & 127;
            out[(long)(i0 + r) * SQ + j0 + c] = NEGV;
        }
        return;
    }
    __shared__ unsigned short Ks[128 * 136];  // k tile, padded stride
    __shared__ unsigned short Qs[64 * 136];
    __shared__ float Ws[64 * 32];
    #pragma unroll
    for (int it = 0; it < 8; ++it) {          // 128x128 halves of k (async)
        int idx = it * 256 + tid;
        int r = idx >> 4, ch = idx & 15;
        async_copy_b128(Ks + r * 136 + ch * 8, Kb + (long)(j0 + r) * 128 + ch * 8);
    }
    #pragma unroll
    for (int it = 0; it < 8; ++it) {          // 64x32 w tile
        int idx = it * 256 + tid;
        Ws[idx] = W[(long)(i0 + (idx >> 5)) * 32 + (idx & 31)];
    }
    int wm = wid >> 2, wn = wid & 3;
    int mB = wm * 32, nB = wn * 32;
    v8f acc[2][2] = {};
    for (int h = 0; h < NH; ++h) {
        __syncthreads();                       // Qs reuse / first-iter W ready
        #pragma unroll
        for (int it = 0; it < 4; ++it) {       // 64x128 halves of q (head h, async)
            int idx = it * 256 + tid;
            int r = idx >> 4, ch = idx & 15;
            async_copy_b128(Qs + r * 136 + ch * 8,
                            Q + (long)(i0 + r) * HD + h * 128 + ch * 8);
        }
        if (h + 1 < NH)
            __builtin_prefetch(Q + (long)(i0 + (tid >> 4)) * HD + (h + 1) * 128, 0, 0);
        wait_async();                          // covers Ks issue too (same wave ctr)
        __syncthreads();
        v8f sc[2][2] = {};
        #pragma unroll
        for (int kk = 0; kk < 4; ++kk) {       // K=128 in 4 chunks of 32
            int ko = kk * 32;
            v16bf a0 = load_frag(Qs, mB, 136, ko);
            v16bf a1 = load_frag(Qs, mB + 16, 136, ko);
            v16bf b0 = load_frag(Ks, nB, 136, ko);
            v16bf b1 = load_frag(Ks, nB + 16, 136, ko);
            sc[0][0] = wmma_bf16(a0, b0, sc[0][0]);
            sc[0][1] = wmma_bf16(a0, b1, sc[0][1]);
            sc[1][0] = wmma_bf16(a1, b0, sc[1][0]);
            sc[1][1] = wmma_bf16(a1, b1, sc[1][1]);
        }
        #pragma unroll
        for (int mt = 0; mt < 2; ++mt) {
            float wv[8];
            #pragma unroll
            for (int g = 0; g < 8; ++g) {
                int row = mB + mt * 16 + g + ((lane >> 4) << 3);
                wv[g] = Ws[row * 32 + h];
            }
            #pragma unroll
            for (int nt = 0; nt < 2; ++nt)
                #pragma unroll
                for (int g = 0; g < 8; ++g)
                    acc[mt][nt][g] += wv[g] * fmaxf(sc[mt][nt][g], 0.0f);
        }
    }
    #pragma unroll
    for (int mt = 0; mt < 2; ++mt)
        #pragma unroll
        for (int nt = 0; nt < 2; ++nt)
            #pragma unroll
            for (int g = 0; g < 8; ++g) {
                int i = i0 + mB + mt * 16 + g + ((lane >> 4) << 3);
                int j = j0 + nB + nt * 16 + (lane & 15);
                out[(long)i * SQ + j] = (j <= i) ? acc[mt][nt][g] : NEGV;
            }
}

// ====== top-k: per-row 2-level radix select + compaction ======
__device__ __forceinline__ unsigned keyf(float f) {
    unsigned b = __float_as_uint(f);
    return b ^ ((b & 0x80000000u) ? 0xFFFFFFFFu : 0x80000000u);
}

__global__ void topk_kernel(const float* __restrict__ logits, int* __restrict__ outIdx) {
    int i = blockIdx.x;
    __shared__ unsigned hist[256];
    __shared__ int sb1, sAll;
    __shared__ unsigned sc1, shi, sc2, ctrA, ctrB;
    int tid = threadIdx.x;
    const float* row = logits + (long)i * SQ;
    int* orow = outIdx + (long)i * TOPK;
    for (int p = tid; p < TOPK; p += 256) orow[p] = -1;
    if (tid < 256) hist[tid] = 0;
    __syncthreads();
    for (int j = tid; j <= i; j += 256)
        atomicAdd(&hist[keyf(row[j]) >> 24], 1u);
    __syncthreads();
    if (tid == 0) {
        unsigned cum = 0; int b1 = -1; unsigned c1 = 0;
        for (int b = 255; b >= 0; --b) {
            if (cum + hist[b] >= TOPK) { b1 = b; c1 = cum; break; }
            cum += hist[b];
        }
        sb1 = b1; sc1 = c1; sAll = (b1 < 0); ctrA = 0;
    }
    __syncthreads();
    if (sAll) {                                  // < 2048 valid: keep everything
        for (int j = tid; j <= i; j += 256) {
            unsigned pos = atomicAdd(&ctrA, 1u);
            orow[pos] = j;
        }
        return;
    }
    int b1 = sb1; unsigned c1 = sc1;
    if (tid < 256) hist[tid] = 0;
    __syncthreads();
    for (int j = tid; j <= i; j += 256) {
        unsigned u = keyf(row[j]);
        if ((u >> 24) == (unsigned)b1) atomicAdd(&hist[(u >> 16) & 255u], 1u);
    }
    __syncthreads();
    if (tid == 0) {
        unsigned cum = c1, hi = 0, c2 = 0;
        for (int b = 255; b >= 0; --b) {
            if (cum + hist[b] >= TOPK) { hi = ((unsigned)b1 << 8) | (unsigned)b; c2 = cum; break; }
            cum += hist[b];
        }
        shi = hi; sc2 = c2; ctrA = 0; ctrB = 0;
    }
    __syncthreads();
    unsigned hi = shi, c2 = sc2;
    for (int j = tid; j <= i; j += 256) {
        unsigned t = keyf(row[j]) >> 16;
        if (t > hi) {
            unsigned pos = atomicAdd(&ctrA, 1u);
            orow[pos] = j;
        } else if (t == hi) {
            unsigned pos = c2 + atomicAdd(&ctrB, 1u);
            if (pos < TOPK) orow[pos] = j;
        }
    }
}

extern "C" void kernel_launch(void* const* d_in, const int* in_sizes, int n_in,
                              void* d_out, int out_size, void* d_ws, size_t ws_size,
                              hipStream_t stream) {
    const float* hidden = (const float*)d_in[0];   // S x HID
    const float* q_lora = (const float*)d_in[1];   // S x QLR
    const int*   pos    = (const int*)d_in[2];     // S
    const float* Wq     = (const float*)d_in[3];   // QLR x HD
    const float* Wk     = (const float*)d_in[4];   // HID x D
    const float* gamma  = (const float*)d_in[5];   // D
    const float* beta   = (const float*)d_in[6];   // D
    const float* Wp     = (const float*)d_in[7];   // HID x H

    // workspace layout (all 16B-aligned by construction)
    unsigned short* hid_bf = (unsigned short*)d_ws;                 // S*HID
    unsigned short* ql_bf  = hid_bf + (long)SQ * HIDN;              // S*QLR
    unsigned short* WqT    = ql_bf + (long)SQ * QLR;                // HD*QLR
    unsigned short* WcT    = WqT + (long)HD * QLR;                  // 160*HID (Wk^T || Wp^T)
    unsigned short* Qbf    = WcT + (long)160 * HIDN;                // S*HD
    unsigned short* Kbf    = Qbf + (long)SQ * HD;                   // S*D
    float*          Wf     = (float*)(Kbf + (long)SQ * DH);         // S*H
    float*          KWf    = Wf + (long)SQ * NH;                    // S*160

    float* logits = (float*)d_out;
    int* topk = (int*)(logits + (long)SQ * SQ);

    long nH = (long)SQ * HIDN, nQ = (long)SQ * QLR;
    cast_f32_bf16<<<(unsigned)((nH + 255) / 256), 256, 0, stream>>>(hidden, hid_bf, nH);
    cast_f32_bf16<<<(unsigned)((nQ + 255) / 256), 256, 0, stream>>>(q_lora, ql_bf, nQ);

    long nWq = (long)QLR * HD;
    transpose_cast<<<(unsigned)((nWq + 255) / 256), 256, 0, stream>>>(Wq, WqT, QLR, HD);
    long nWk = (long)HIDN * DH;
    transpose_cast<<<(unsigned)((nWk + 255) / 256), 256, 0, stream>>>(Wk, WcT, HIDN, DH);
    long nWp = (long)HIDN * NH;
    transpose_cast<<<(unsigned)((nWp + 255) / 256), 256, 0, stream>>>(
        Wp, WcT + (long)DH * HIDN, HIDN, NH);

    // kw = hidden @ [Wk | Wp]  (N = 160)
    gemm_bf16_nt<<<dim3(3, SQ / 64), 256, 0, stream>>>(hid_bf, WcT, KWf, SQ, 160, HIDN);
    kw_post<<<SQ / 8, 256, 0, stream>>>(KWf, gamma, beta, pos, Kbf, Wf);

    // q = q_lora @ Wq, fused RoPE + FWHT per head
    qgemm_rope_fwht<<<dim3(NH, SQ / 64), 256, 0, stream>>>(ql_bf, WqT, pos, Qbf);

    // logits + causal mask
    logits_kernel<<<dim3(SQ / 128, SQ / 64), 256, 0, stream>>>(Qbf, Kbf, Wf, logits);

    // per-row top-2048
    topk_kernel<<<SQ, 256, 0, stream>>>(logits, topk);
}
